// GPe_25434796327643
// MI455X (gfx1250) — matrix operand order repaired and checked by reading, not compiled, not run on hardware
//
#include <hip/hip_runtime.h>
#include <hip/hip_bf16.h>

// out[16384,3072] = inputs[16384,3072] @ (w * mask^T)[3072,3072] + bias
// Compute-bound at bf16 WMMA rate (309 GFLOP vs ~21us of HBM traffic), so:
//   pass 1: pack inputs f32->bf16 in WMMA A-fragment order
//   pass 2: pack (w * mask^T) f32->bf16 in WMMA B-fragment order
//   pass 3: GEMM: 8 wave32/WG, 256x128 WG tile, 64x64 per wave (4x4 accums),
//           16 v_wmma_f32_16x16x32_bf16 per 8 coalesced b128-pair frag loads.

#define BATCH   16384
#define IN_DIM  3072
#define UNITS   3072

#define KBLK (IN_DIM / 32)   // 96  K-blocks of 32
#define MBLK (BATCH / 16)    // 1024 M-blocks of 16
#define NBLK (UNITS / 16)    // 192  N-blocks of 16
// One fragment = 512 bf16 = 1024 bytes = 32 lanes x 16 bf16 (32B per lane)

typedef __attribute__((ext_vector_type(16))) __bf16 v16bf;
typedef __attribute__((ext_vector_type(8)))  float  v8f;

__device__ __forceinline__ unsigned short f32_to_bf16_rne(float f) {
    unsigned u = __builtin_bit_cast(unsigned, f);
    unsigned r = u + 0x7FFFu + ((u >> 16) & 1u);   // round-to-nearest-even
    return (unsigned short)(r >> 16);
}

// ---------------------------------------------------------------------------
// Pass 1: pack A. A 16x32 fragment, contiguous k order used here (matches B; wave32):
//   lane L<16 : row = L,    K = 0..15   (VGPR v holds K pair 2v,2v+1)
//   lane L>=16: row = L-16, K = 16..31
// Packed index: ((mblk*KBLK + kblk)*32 + lane)*16 + e
//   k = kblk*32 + e + (lane>=16 ? 16 : 0)
// One thread emits one dword (bf16 pair e,e+1 = consecutive k).
// ---------------------------------------------------------------------------
__global__ __launch_bounds__(256) void pack_a_kernel(const float* __restrict__ A,
                                                     unsigned* __restrict__ Ap) {
    const int gid  = blockIdx.x * 256 + threadIdx.x;
    const int ep   = gid & 7;
    const int lane = (gid >> 3) & 31;
    const int frag = gid >> 8;                         // = mblk*KBLK + kblk
    const int kb   = frag % KBLK;
    const int mb   = frag / KBLK;
    const int e    = ep << 1;
    const int m    = mb * 16 + (lane & 15);
    const int k    = kb * 32 + e + ((lane >> 4) << 4);
    const float2 v = *(const float2*)(A + (size_t)m * IN_DIM + k);
    Ap[gid] = (unsigned)f32_to_bf16_rne(v.x) | ((unsigned)f32_to_bf16_rne(v.y) << 16);
}

// ---------------------------------------------------------------------------
// Pass 2: pack B = w * mask^T. B 32x16 fragment, contiguous k order used here (matches A; wave32):
//   lanes 0..15 : col n = lane,    K = 0..15 (VGPR v holds K pair 2v,2v+1)
//   lanes 16..31: col n = lane-16, K = 16..31
// Packed index: ((kblk*NBLK + nblk)*32 + lane)*16 + e
//   k = kblk*32 + e + (lane>=16 ? 16 : 0);  n = nblk*16 + (lane&15)
// ---------------------------------------------------------------------------
__global__ __launch_bounds__(256) void pack_b_kernel(const float* __restrict__ W,
                                                     const int* __restrict__ Mask,
                                                     unsigned* __restrict__ Bp) {
    const int gid  = blockIdx.x * 256 + threadIdx.x;
    const int ep   = gid & 7;
    const int lane = (gid >> 3) & 31;
    const int frag = gid >> 8;                         // = kblk*NBLK + nblk
    const int nb   = frag % NBLK;
    const int kb   = frag / NBLK;
    const int e    = ep << 1;
    const int n    = nb * 16 + (lane & 15);
    const int k    = kb * 32 + e + ((lane >> 4) << 4);
    const float w0 = W[(size_t)k * UNITS + n]       * (float)Mask[(size_t)n * IN_DIM + k];
    const float w1 = W[(size_t)(k + 1) * UNITS + n] * (float)Mask[(size_t)n * IN_DIM + k + 1];
    Bp[gid] = (unsigned)f32_to_bf16_rne(w0) | ((unsigned)f32_to_bf16_rne(w1) << 16);
}

// ---------------------------------------------------------------------------
// Pass 3: GEMM. 256 threads = 8 wave32. WG tile 256x128, wave tile 64x64.
// Wave grid: wm = wave>>1 (4 row blocks of 64), wn = wave&1 (2 col blocks of 64).
// Per K-step per wave: 8 fragment loads (16 b128, all offsets fold into the
// 24-bit imm: A subtile stride 98304B, B subtile stride 1024B) + 16 WMMA.
// Running pointers advance by constants -> minimal in-loop VALU.
// ---------------------------------------------------------------------------
__global__ __launch_bounds__(256) void gemm_wmma_kernel(const v16bf* __restrict__ Af,
                                                        const v16bf* __restrict__ Bf,
                                                        const float* __restrict__ bias,
                                                        float* __restrict__ out) {
    const int lane = threadIdx.x & 31;
    const int wave = threadIdx.x >> 5;
    const int wm   = wave >> 1;
    const int wn   = wave & 1;
    const int m0   = blockIdx.x * 256 + wm * 64;   // 4 x 16-row subtiles
    const int n0   = blockIdx.y * 128 + wn * 64;   // 4 x 16-col subtiles

    v8f acc[4][4] = {};

    // Fragment-granular running pointers (frag = 32 v16bf elements).
    const v16bf* ap = Af + (size_t)(m0 >> 4) * (KBLK * 32) + lane;
    const v16bf* bp = Bf + (size_t)(n0 >> 4) * 32 + lane;

    for (int kb = 0; kb < KBLK; ++kb) {
        v16bf a[4], b[4];
#pragma unroll
        for (int i = 0; i < 4; ++i)
            a[i] = ap[i * (KBLK * 32)];            // +0 / 98304B / 196608B / 294912B
#pragma unroll
        for (int j = 0; j < 4; ++j)
            b[j] = bp[j * 32];                     // +0 / 1024B / 2048B / 3072B
#pragma unroll
        for (int i = 0; i < 4; ++i)
#pragma unroll
            for (int j = 0; j < 4; ++j)
                acc[i][j] = __builtin_amdgcn_wmma_f32_16x16x32_bf16(
                    /*neg_a=*/false, a[i], /*neg_b=*/false, b[j],
                    /*c_mod=*/(short)0, acc[i][j],
                    /*reuse_a=*/false, /*reuse_b=*/false);
        ap += 32;              // next K-block: +1024B
        bp += NBLK * 32;       // next K-block: +196608B
    }

    // Epilogue. C/D layout: VGPR r -> row r (lanes 0-15) / row r+8 (lanes 16-31),
    // lane&15 -> column within the 16-wide tile. Fuse bias.
    const int col_in = lane & 15;
    const int rowoff = (lane >> 4) << 3;
#pragma unroll
    for (int j = 0; j < 4; ++j) {
        const int col = n0 + j * 16 + col_in;
        const float bv = bias[col];
#pragma unroll
        for (int i = 0; i < 4; ++i) {
            const int rbase = m0 + i * 16 + rowoff;
#pragma unroll
            for (int r = 0; r < 8; ++r)
                out[(size_t)(rbase + r) * UNITS + col] = acc[i][j][r] + bv;
        }
    }
}

extern "C" void kernel_launch(void* const* d_in, const int* in_sizes, int n_in,
                              void* d_out, int out_size, void* d_ws, size_t ws_size,
                              hipStream_t stream) {
    const float* inputs = (const float*)d_in[0];   // [BATCH, IN_DIM] f32
    const float* w      = (const float*)d_in[1];   // [IN_DIM, UNITS] f32
    const float* bias   = (const float*)d_in[2];   // [UNITS] f32
    const int*   mask   = (const int*)d_in[3];     // [UNITS, IN_DIM] i32
    float*       out    = (float*)d_out;           // [BATCH, UNITS] f32

    // Workspace: packed A (100.7 MB) then packed B (18.9 MB); both L2-resident.
    const size_t a_frags = (size_t)MBLK * KBLK;    // 98304
    const size_t b_frags = (size_t)KBLK * NBLK;    // 18432
    unsigned* Ap = (unsigned*)d_ws;
    unsigned* Bp = Ap + a_frags * 256;

    pack_a_kernel<<<dim3((unsigned)a_frags), 256, 0, stream>>>(inputs, Ap);
    pack_b_kernel<<<dim3((unsigned)b_frags), 256, 0, stream>>>(w, mask, Bp);
    gemm_wmma_kernel<<<dim3(BATCH / 256, UNITS / 128), 256, 0, stream>>>(
        (const v16bf*)Ap, (const v16bf*)Bp, bias, out);
}
